// MoELayer_16149077033149
// MI455X (gfx1250) — matrix operand order repaired and checked by reading, not compiled, not run on hardware
//
#include <hip/hip_runtime.h>
#include <hip/hip_bf16.h>

// MoE top-2 dispatch + fused expert FFN for MI455X (gfx1250, wave32, WMMA).
// N=8192, D=1024, H=4096, E=8, k=2.
// Top-2 routing is math-equivalent to the dense reference (dispatch_mask
// zeroes all other experts): 275 GFLOP instead of 1.1 TFLOP.
// 64-token M-tiles: weight traffic ~8.6 GB, mostly served from the 192 MB L2
// because same-expert tiles are grid-adjacent; compute-bound on bf16 WMMA.

#define MOE_N 8192
#define MOE_D 1024
#define MOE_H 4096
#define MOE_E 8

#define TILE_M 64        // tokens per block
#define SM     4         // 16-row M subtiles per block (TILE_M/16)

#define XS_STRIDE 1032   // 1024 + 8 halfs pad: row stride 516 dwords -> banks rotate by 4
#define HS_STRIDE 136    // 128 + 8 halfs pad

typedef __attribute__((ext_vector_type(16))) __bf16 v16bf;
typedef __attribute__((ext_vector_type(8)))  float  v8f;

union ABf { v16bf v; unsigned int p[8]; };

// Pack two f32 into two bf16 (truncation) with a single v_perm_b32.
__device__ __forceinline__ unsigned int pack_bf2(float lo, float hi) {
  return __builtin_amdgcn_perm(__float_as_uint(hi), __float_as_uint(lo), 0x07060302u);
}

// ---------------------------------------------------------------------------
// Kernel 0: zero output accumulator + routing counters.
// ---------------------------------------------------------------------------
__global__ void moe_zero(float* __restrict__ out, int* __restrict__ counts,
                         float* __restrict__ importance, long long n) {
  long long i = (long long)blockIdx.x * blockDim.x + threadIdx.x;
  if (i < n) out[i] = 0.f;
  if (i < MOE_E) { counts[i] = 0; importance[i] = 0.f; }
}

// ---------------------------------------------------------------------------
// Kernel 1: router. One wave (32 lanes) per token: logits = x@Wr + br,
// softmax over E=8, top-2 (strict '>' scan == JAX lowest-index tie-break),
// scatter token into per-expert lists, accumulate importance.
// ---------------------------------------------------------------------------
__global__ __launch_bounds__(256) void moe_router(
    const float* __restrict__ x, const float* __restrict__ Wr,
    const float* __restrict__ br, int* __restrict__ counts,
    float* __restrict__ importance, int* __restrict__ tlist,
    float* __restrict__ glist) {
  const int wave = threadIdx.x >> 5, lane = threadIdx.x & 31;
  const int n = blockIdx.x * 8 + wave;
  float acc[MOE_E] = {0.f,0.f,0.f,0.f,0.f,0.f,0.f,0.f};
  const float* xr = x + (size_t)n * MOE_D;
  for (int d = lane; d < MOE_D; d += 32) {       // coalesced: lane-consecutive
    float xv = xr[d];
    const float4* w = (const float4*)(Wr + d * MOE_E);  // 32B row per d
    float4 w0 = w[0], w1 = w[1];
    acc[0] += xv * w0.x; acc[1] += xv * w0.y; acc[2] += xv * w0.z; acc[3] += xv * w0.w;
    acc[4] += xv * w1.x; acc[5] += xv * w1.y; acc[6] += xv * w1.z; acc[7] += xv * w1.w;
  }
  for (int e = 0; e < MOE_E; ++e)
    for (int off = 16; off > 0; off >>= 1)
      acc[e] += __shfl_xor(acc[e], off, 32);
  if (lane == 0) {
    float l[MOE_E], m = -1e30f;
    for (int e = 0; e < MOE_E; ++e) { l[e] = acc[e] + br[e]; m = fmaxf(m, l[e]); }
    float s = 0.f;
    for (int e = 0; e < MOE_E; ++e) { l[e] = __expf(l[e] - m); s += l[e]; }
    float inv = 1.f / s;
    int i0 = -1, i1 = -1; float v0 = -1.f, v1 = -1.f;
    for (int e = 0; e < MOE_E; ++e) {
      float p = l[e] * inv;
      if (p > v0)      { v1 = v0; i1 = i0; v0 = p; i0 = e; }
      else if (p > v1) { v1 = p;  i1 = e; }
    }
    int p0 = atomicAdd(&counts[i0], 1);
    tlist[i0 * MOE_N + p0] = n; glist[i0 * MOE_N + p0] = v0;
    atomicAdd(&importance[i0], v0);
    int p1 = atomicAdd(&counts[i1], 1);
    tlist[i1 * MOE_N + p1] = n; glist[i1 * MOE_N + p1] = v1;
    atomicAdd(&importance[i1], v1);
  }
}

// ---------------------------------------------------------------------------
// Kernel 2: aux loss = var(importance, ddof=1) / (mean^2 + 1e-9).
// ---------------------------------------------------------------------------
__global__ void moe_aux(const float* __restrict__ importance, float* __restrict__ aux_out) {
  if (threadIdx.x == 0 && blockIdx.x == 0) {
    float s = 0.f;
    for (int e = 0; e < MOE_E; ++e) s += importance[e];
    float mean = s / (float)MOE_E, v = 0.f;
    for (int e = 0; e < MOE_E; ++e) { float d = importance[e] - mean; v += d * d; }
    v /= (float)(MOE_E - 1);
    *aux_out = v / (mean * mean + 1e-9f);
  }
}

// ---------------------------------------------------------------------------
// Kernel 3: fused expert FFN. Block = (expert e, 64-token tile), 8 waves.
// Per 128-wide H chunk:
//   Phase A: wave w computes h[0:64, 16w:16w+16] over K=1024
//            (4 M-subtiles x 32 k-steps; B fragment reused 4x),
//            ReLU(h+b1) -> LDS bf16.
//   Phase B: wave w accumulates out[0:64, 128w:128w+128]
//            (4 M x 8 N subtiles = 32 v8f accumulators; A reused 8x, B 4x).
// Epilogue: gate * (y + b2) atomically added into zeroed output (exactly 2
// commutative f32 adds per element -> deterministic).
// ---------------------------------------------------------------------------
__global__ __launch_bounds__(256) void moe_ffn(
    const float* __restrict__ x, const float* __restrict__ W1,
    const float* __restrict__ b1, const float* __restrict__ W2,
    const float* __restrict__ b2, const int* __restrict__ counts,
    const int* __restrict__ tlist, const float* __restrict__ glist,
    float* __restrict__ out) {
  extern __shared__ unsigned short smem[];
  unsigned short* Xs = smem;                          // TILE_M x 1024 bf16
  unsigned short* Hs = smem + TILE_M * XS_STRIDE;     // TILE_M x 128  bf16

  const int e   = blockIdx.x;
  const int cnt = counts[e];
  const int m0  = blockIdx.y * TILE_M;
  if (m0 >= cnt) return;                              // uniform across block
  const int rows = min(TILE_M, cnt - m0);
  const int wave = threadIdx.x >> 5, lane = threadIdx.x & 31;
  const int rlo = lane & 15;                          // A row / C col
  const int hi8 = (lane >> 4) * 8;                    // K-phase offset (ISA A layout)

  // Stage x tile: fp32 -> bf16 (truncate) into LDS; pad rows with zeros.
  for (int i = threadIdx.x; i < TILE_M * MOE_D; i += 256) {
    int r = i >> 10, c = i & (MOE_D - 1);
    float v = 0.f;
    if (r < rows) v = x[(size_t)tlist[e * MOE_N + m0 + r] * MOE_D + c];
    Xs[r * XS_STRIDE + c] = (unsigned short)(__float_as_uint(v) >> 16);
  }
  __syncthreads();

  v8f c2[SM * 8];
  { v8f z = {}; for (int i = 0; i < SM * 8; ++i) c2[i] = z; }

  for (int hc = 0; hc < MOE_H / 128; ++hc) {
    const int hb = hc * 128 + wave * 16;
    v8f c1[SM];
    { v8f z = {}; for (int m = 0; m < SM; ++m) c1[m] = z; }
    const float* w1base = W1 + (size_t)e * MOE_D * MOE_H + hb;
    for (int k0 = 0; k0 < MOE_D; k0 += 32) {
      // B fragment: lane = H column; K rows hi8+0..7, 16+hi8+0..7, fp32 -> bf16.
      const float* bp = w1base + (size_t)(k0 + hi8) * MOE_H + rlo;
      float4 f0 = make_float4(bp[0], bp[(size_t)1 * MOE_H], bp[(size_t)2 * MOE_H], bp[(size_t)3 * MOE_H]);
      float4 f1 = make_float4(bp[(size_t)4 * MOE_H], bp[(size_t)5 * MOE_H], bp[(size_t)6 * MOE_H], bp[(size_t)7 * MOE_H]);
      float4 f2 = make_float4(bp[(size_t)16 * MOE_H], bp[(size_t)17 * MOE_H], bp[(size_t)18 * MOE_H], bp[(size_t)19 * MOE_H]);
      float4 f3 = make_float4(bp[(size_t)20 * MOE_H], bp[(size_t)21 * MOE_H], bp[(size_t)22 * MOE_H], bp[(size_t)23 * MOE_H]);
      if (k0 + 32 < MOE_D) __builtin_prefetch(bp + (size_t)32 * MOE_H, 0, 3);
      ABf b;
      b.p[0] = pack_bf2(f0.x, f0.y); b.p[1] = pack_bf2(f0.z, f0.w);
      b.p[2] = pack_bf2(f1.x, f1.y); b.p[3] = pack_bf2(f1.z, f1.w);
      b.p[4] = pack_bf2(f2.x, f2.y); b.p[5] = pack_bf2(f2.z, f2.w);
      b.p[6] = pack_bf2(f3.x, f3.y); b.p[7] = pack_bf2(f3.z, f3.w);
#pragma unroll
      for (int m = 0; m < SM; ++m) {
        // A fragment: 2 x ds_load_b128 (16-bit 16x32 A layout).
        ABf a;
        const unsigned short* xrow = &Xs[(m * 16 + rlo) * XS_STRIDE + k0 + hi8];
        *(uint4*)&a.p[0] = *(const uint4*)xrow;
        *(uint4*)&a.p[4] = *(const uint4*)(xrow + 16);
        c1[m] = __builtin_amdgcn_wmma_f32_16x16x32_bf16(false, a.v, false, b.v,
                                                        (short)0, c1[m], false, false);
      }
    }
    // ReLU(h + b1) -> LDS as bf16. C layout: VGPR j -> row j + 8*(lane>=16).
    {
      float bias = b1[e * MOE_H + hb + rlo];
#pragma unroll
      for (int m = 0; m < SM; ++m)
#pragma unroll
        for (int j = 0; j < 8; ++j) {
          float v = fmaxf(c1[m][j] + bias, 0.f);
          Hs[(m * 16 + j + hi8) * HS_STRIDE + wave * 16 + rlo] =
              (unsigned short)(__float_as_uint(v) >> 16);
        }
    }
    __syncthreads();

    // Phase B: out[0:64, 128w..] += h_chunk @ W2[e][hc*128.., 128w..]
    const float* w2base = W2 + ((size_t)e * MOE_H + hc * 128) * MOE_D + wave * 128;
    for (int k0 = 0; k0 < 128; k0 += 32) {
      ABf a[SM];
#pragma unroll
      for (int m = 0; m < SM; ++m) {
        const unsigned short* hrow = &Hs[(m * 16 + rlo) * HS_STRIDE + k0 + hi8];
        *(uint4*)&a[m].p[0] = *(const uint4*)hrow;
        *(uint4*)&a[m].p[4] = *(const uint4*)(hrow + 16);
      }
      const float* brow = w2base + (size_t)(k0 + hi8) * MOE_D + rlo;
      if (k0 + 32 < 128) __builtin_prefetch(brow + (size_t)32 * MOE_D, 0, 3);
#pragma unroll
      for (int nt = 0; nt < 8; ++nt) {
        const float* bp = brow + nt * 16;
        float4 f0 = make_float4(bp[0], bp[(size_t)1 * MOE_D], bp[(size_t)2 * MOE_D], bp[(size_t)3 * MOE_D]);
        float4 f1 = make_float4(bp[(size_t)4 * MOE_D], bp[(size_t)5 * MOE_D], bp[(size_t)6 * MOE_D], bp[(size_t)7 * MOE_D]);
        float4 f2 = make_float4(bp[(size_t)16 * MOE_D], bp[(size_t)17 * MOE_D], bp[(size_t)18 * MOE_D], bp[(size_t)19 * MOE_D]);
        float4 f3 = make_float4(bp[(size_t)20 * MOE_D], bp[(size_t)21 * MOE_D], bp[(size_t)22 * MOE_D], bp[(size_t)23 * MOE_D]);
        ABf b;
        b.p[0] = pack_bf2(f0.x, f0.y); b.p[1] = pack_bf2(f0.z, f0.w);
        b.p[2] = pack_bf2(f1.x, f1.y); b.p[3] = pack_bf2(f1.z, f1.w);
        b.p[4] = pack_bf2(f2.x, f2.y); b.p[5] = pack_bf2(f2.z, f2.w);
        b.p[6] = pack_bf2(f3.x, f3.y); b.p[7] = pack_bf2(f3.z, f3.w);
#pragma unroll
        for (int m = 0; m < SM; ++m)
          c2[nt * SM + m] = __builtin_amdgcn_wmma_f32_16x16x32_bf16(
              false, a[m].v, false, b.v, (short)0, c2[nt * SM + m], false, false);
      }
    }
    __syncthreads();
  }

  // Epilogue: gate * (y + b2) scattered into output.
  float b2v[8];
#pragma unroll
  for (int nt = 0; nt < 8; ++nt)
    b2v[nt] = b2[e * MOE_D + wave * 128 + nt * 16 + rlo];
#pragma unroll
  for (int m = 0; m < SM; ++m)
#pragma unroll
    for (int j = 0; j < 8; ++j) {
      int M = m * 16 + j + hi8;
      if (M < rows) {
        int   tok = tlist[e * MOE_N + m0 + M];
        float g   = glist[e * MOE_N + m0 + M];
        float* orow = out + (size_t)tok * MOE_D + wave * 128 + rlo;
#pragma unroll
        for (int nt = 0; nt < 8; ++nt)
          atomicAdd(orow + nt * 16, g * (c2[nt * SM + m][j] + b2v[nt]));
      }
    }
}

// ---------------------------------------------------------------------------
// Launcher. Workspace layout (needs ~0.52 MB):
//   int   counts[8]; float importance[8]; int tlist[E*N]; float glist[E*N];
// ---------------------------------------------------------------------------
extern "C" void kernel_launch(void* const* d_in, const int* in_sizes, int n_in,
                              void* d_out, int out_size, void* d_ws, size_t ws_size,
                              hipStream_t stream) {
  const float* x  = (const float*)d_in[0];
  const float* Wr = (const float*)d_in[1];
  const float* br = (const float*)d_in[2];
  const float* W1 = (const float*)d_in[3];
  const float* b1 = (const float*)d_in[4];
  const float* W2 = (const float*)d_in[5];
  const float* b2 = (const float*)d_in[6];
  // d_in[7] is k == 2 (baked into the kernels).
  float* out = (float*)d_out;

  int*   counts     = (int*)d_ws;
  float* importance = (float*)d_ws + 8;
  int*   tlist      = (int*)d_ws + 16;
  float* glist      = (float*)d_ws + 16 + MOE_E * MOE_N;

  // Dynamic LDS: 64x1024 bf16 x-tile + 64x128 bf16 h-chunk (~146 KB; WGP has 320 KB).
  const size_t lds_bytes =
      (size_t)(TILE_M * XS_STRIDE + TILE_M * HS_STRIDE) * sizeof(unsigned short);
  static bool attr_set = false;
  if (!attr_set) {   // host-side attribute, not a stream op; safe under capture
    (void)hipFuncSetAttribute((const void*)moe_ffn,
                              hipFuncAttributeMaxDynamicSharedMemorySize,
                              (int)lds_bytes);
    attr_set = true;
  }

  const long long nout = (long long)MOE_N * MOE_D;
  moe_zero<<<(unsigned)((nout + 255) / 256), 256, 0, stream>>>(out, counts, importance, nout);
  moe_router<<<MOE_N / 8, 256, 0, stream>>>(x, Wr, br, counts, importance, tlist, glist);
  moe_aux<<<1, 32, 0, stream>>>(importance, out + (size_t)MOE_N * MOE_D);
  // Worst case: one expert receives all N tokens -> N/TILE_M tiles; surplus
  // tiles early-exit on counts[e].
  moe_ffn<<<dim3(MOE_E, MOE_N / TILE_M), 256, lds_bytes, stream>>>(
      x, W1, b1, W2, b2, counts, tlist, glist, out);
}
